// ResidualVectorQuantizer_74156905332759
// MI455X (gfx1250) — matrix operand order repaired and checked by reading, not compiled, not run on hardware
//
#include <hip/hip_runtime.h>
#include <math.h>

#define N_PTS     131072
#define DIM       128
#define LEVELS    3
#define KCODES    1024
#define BETA      0.25f
#define USAGE_REGC 1e-3f
#define ROWS_PER_BLOCK 16
#define A_PAD     136   // 128 + 8 bf16 pad -> 272B row stride, conflict-free frag loads

typedef __attribute__((ext_vector_type(16))) __bf16 v16bf;
typedef __attribute__((ext_vector_type(8)))  __bf16 v8bf;
typedef __attribute__((ext_vector_type(8)))  float  v8f;

static __device__ __forceinline__ __bf16 f32_to_bf16(float f) {
  union { float f; unsigned u; } a; a.f = f;
  unsigned r = a.u + 0x7FFFu + ((a.u >> 16) & 1u);   // round-to-nearest-even
  unsigned short h = (unsigned short)(r >> 16);
  return __builtin_bit_cast(__bf16, h);
}

// A fragment, 16x32 bf16 (M x K): lanes 0-15 = rows, hold K {0..7,16..23};
// lanes 16-31 = same rows, K {8..15,24..31}  (ISA 7.12.2 table)
static __device__ __forceinline__ v16bf load_a_frag(const __bf16* p) {
  v8bf lo = *reinterpret_cast<const v8bf*>(p);        // 8 contiguous bf16
  v8bf hi = *reinterpret_cast<const v8bf*>(p + 16);   // +16 features
  v16bf r;
#pragma unroll
  for (int i = 0; i < 8; ++i) { r[i] = lo[i]; r[i + 8] = hi[i]; }
  return r;
}

// B fragment, 32x16 bf16 (K x N): lanes 0-15 hold K {0..7,16..23}; lanes 16-31 hold K {8..15,24..31}
static __device__ __forceinline__ v16bf load_b_frag(const __bf16* p) {
  v8bf lo = *reinterpret_cast<const v8bf*>(p);
  v8bf hi = *reinterpret_cast<const v8bf*>(p + 16);
  v16bf r;
#pragma unroll
  for (int i = 0; i < 8; ++i) { r[i] = lo[i]; r[i + 8] = hi[i]; }
  return r;
}

__global__ void rvq_init(float* __restrict__ probs_acc, float* __restrict__ commit_sum) {
  int t = threadIdx.x;
  for (int i = t; i < LEVELS * KCODES; i += blockDim.x) probs_acc[i] = 0.f;
  if (t == 0) commit_sum[0] = 0.f;
}

// Convert codebooks f32->bf16, compute ||W_k||^2.  One wave per codebook row.
__global__ void rvq_prep(const float* __restrict__ cb,
                         __bf16* __restrict__ cb16,
                         float* __restrict__ wnorm) {
  int lane = threadIdx.x & 31;
  int row  = blockIdx.x * 8 + (threadIdx.x >> 5);          // 3*1024 rows total
  if (row >= LEVELS * KCODES) return;
  const float* wr = cb + (size_t)row * DIM;
  float4 v = reinterpret_cast<const float4*>(wr)[lane];    // 4 f32 per lane
  __bf16* dst = cb16 + (size_t)row * DIM + lane * 4;
  dst[0] = f32_to_bf16(v.x); dst[1] = f32_to_bf16(v.y);
  dst[2] = f32_to_bf16(v.z); dst[3] = f32_to_bf16(v.w);
  float s = v.x * v.x + v.y * v.y + v.z * v.z + v.w * v.w;
#pragma unroll
  for (int off = 16; off > 0; off >>= 1) s += __shfl_xor(s, off, 32);
  if (lane == 0) wnorm[row] = s;
}

// One RVQ level: 16 rows per block, distances to all 1024 codes via bf16 WMMA.
__global__ void __launch_bounds__(256)
rvq_level(const float* __restrict__ src,        // residual in (level 0: x)
          float* __restrict__ dst,              // residual out (ws)
          const __bf16* __restrict__ cb16,      // [K][DIM] this level, bf16
          const float* __restrict__ wnorm,      // [K] this level
          const float* __restrict__ cbf32,      // [K][DIM] this level, f32
          float* __restrict__ probs_acc,        // [K] this level
          float* __restrict__ commit_sum,       // scalar
          float* __restrict__ qsum,             // d_out [N][DIM]
          float* __restrict__ codes,            // d_out codes [N][LEVELS]
          int level, int is_first) {
  __shared__ __bf16 A_lds[ROWS_PER_BLOCK][A_PAD];
  __shared__ float  D_lds[ROWS_PER_BLOCK][KCODES];   // 64 KB distance tile
  __shared__ float  probs_lds[KCODES];
  __shared__ float  xnorm_lds[ROWS_PER_BLOCK];
  __shared__ int    idx_lds[ROWS_PER_BLOCK];

  const int tid  = threadIdx.x;
  const int lane = tid & 31;
  const int wave = tid >> 5;                     // 8 waves
  const int l    = lane & 15;
  const int half = lane >> 4;
  const size_t nbase = (size_t)blockIdx.x * ROWS_PER_BLOCK;

  for (int i = tid; i < KCODES; i += 256) probs_lds[i] = 0.f;
  if (tid < ROWS_PER_BLOCK) xnorm_lds[tid] = 0.f;
  __syncthreads();

  // ---- load 16 residual rows: f32 -> bf16 into LDS, accumulate row norms
  for (int e = tid; e < ROWS_PER_BLOCK * DIM; e += 256) {
    int r = e >> 7, c = e & (DIM - 1);
    float v = src[(nbase + r) * DIM + c];
    A_lds[r][c] = f32_to_bf16(v);
    atomicAdd(&xnorm_lds[r], v * v);
  }
  __syncthreads();

  // ---- phase 1: distance tile via WMMA bf16 (f32 accumulate)
  v16bf afrag[4];
#pragma unroll
  for (int kk = 0; kk < 4; ++kk)
    afrag[kk] = load_a_frag(&A_lds[l][kk * 32 + half * 8]);

#pragma unroll 1
  for (int j = 0; j < 8; ++j) {
    const int c0 = (wave * 8 + j) * 16;          // column tile base
    v8f acc = {};
#pragma unroll
    for (int kk = 0; kk < 4; ++kk) {
      v16bf bfrag = load_b_frag(cb16 + (size_t)(c0 + l) * DIM + kk * 32 + half * 8);
      acc = __builtin_amdgcn_wmma_f32_16x16x32_bf16(
          false, afrag[kk], false, bfrag, (short)0, acc, false, false);
    }
    const float wn = wnorm[c0 + l];
#pragma unroll
    for (int r = 0; r < 8; ++r) {
      int m = r + half * 8;
      D_lds[m][c0 + l] = xnorm_lds[m] - 2.0f * acc[r] + wn;
    }
  }
  __syncthreads();

  // ---- phase 2: per-row argmin + stable softmax column-sums (2 rows / wave)
#pragma unroll
  for (int rr = 0; rr < 2; ++rr) {
    const int row = wave + rr * 8;
    const float* drow = D_lds[row];
    float dmin = 3.4e38f; int amin = 0;
    for (int k = lane; k < KCODES; k += 32) {
      float v = drow[k];
      if (v < dmin) { dmin = v; amin = k; }
    }
#pragma unroll
    for (int off = 16; off > 0; off >>= 1) {
      float od = __shfl_xor(dmin, off, 32);
      int   oi = __shfl_xor(amin, off, 32);
      if (od < dmin || (od == dmin && oi < amin)) { dmin = od; amin = oi; }
    }
    float ssum = 0.f;
    for (int k = lane; k < KCODES; k += 32) ssum += __expf(dmin - drow[k]);
#pragma unroll
    for (int off = 16; off > 0; off >>= 1) ssum += __shfl_xor(ssum, off, 32);
    const float inv = 1.0f / ssum;
    for (int k = lane; k < KCODES; k += 32)
      atomicAdd(&probs_lds[k], __expf(dmin - drow[k]) * inv);
    if (lane == 0) {
      idx_lds[row] = amin;
      codes[(nbase + row) * LEVELS + level] = (float)amin;
    }
  }
  __syncthreads();

  // ---- phase 3: gather q (f32), residual update, qsum, commit, flush probs
  float csum = 0.f;
  for (int e = tid; e < ROWS_PER_BLOCK * DIM; e += 256) {
    int r = e >> 7, c = e & (DIM - 1);
    size_t g = (nbase + r) * DIM + c;
    float rv = src[g];
    float q  = cbf32[(size_t)idx_lds[r] * DIM + c];
    float nr = rv - q;
    dst[g] = nr;
    if (is_first) qsum[g] = q; else qsum[g] += q;
    csum += nr * nr;
  }
#pragma unroll
  for (int off = 16; off > 0; off >>= 1) csum += __shfl_xor(csum, off, 32);
  if (lane == 0) atomicAdd(commit_sum, csum);
  for (int k = tid; k < KCODES; k += 256) atomicAdd(&probs_acc[k], probs_lds[k]);
}

__global__ void rvq_final(const float* __restrict__ probs_acc,
                          const float* __restrict__ commit_sum,
                          float* __restrict__ loss_out) {
  __shared__ float red[1024];
  int t = threadIdx.x;
  float term = 0.f;
#pragma unroll
  for (int lv = 0; lv < LEVELS; ++lv) {
    float avg = probs_acc[lv * KCODES + t] * (1.0f / (float)N_PTS);
    term += avg * logf(avg * (float)KCODES + 1e-8f);
  }
  red[t] = term;
  __syncthreads();
  for (int s = 512; s > 0; s >>= 1) {
    if (t < s) red[t] += red[t + s];
    __syncthreads();
  }
  if (t == 0) {
    loss_out[0] = commit_sum[0] * (1.0f + BETA) / ((float)N_PTS * (float)DIM);
    loss_out[1] = USAGE_REGC * red[0];
  }
}

extern "C" void kernel_launch(void* const* d_in, const int* in_sizes, int n_in,
                              void* d_out, int out_size, void* d_ws, size_t ws_size,
                              hipStream_t stream) {
  (void)in_sizes; (void)n_in; (void)out_size; (void)ws_size;
  const float* x  = (const float*)d_in[0];
  const float* cb = (const float*)d_in[1];              // [3][1024][128] f32
  float* out = (float*)d_out;

  char* ws = (char*)d_ws;
  float*  residual = (float*)ws;
  size_t  off = (size_t)N_PTS * DIM * sizeof(float);
  __bf16* cb16 = (__bf16*)(ws + off);  off += (size_t)LEVELS * KCODES * DIM * sizeof(__bf16);
  float*  wnorm = (float*)(ws + off);  off += (size_t)LEVELS * KCODES * sizeof(float);
  float*  probs = (float*)(ws + off);  off += (size_t)LEVELS * KCODES * sizeof(float);
  float*  commit = (float*)(ws + off);

  float* qsum  = out;                                   // [N][DIM]
  float* codes = out + (size_t)N_PTS * DIM;             // [N][LEVELS]
  float* lossp = codes + (size_t)N_PTS * LEVELS;        // [commit, usage]

  rvq_init<<<1, 256, 0, stream>>>(probs, commit);
  rvq_prep<<<(LEVELS * KCODES) / 8, 256, 0, stream>>>(cb, cb16, wnorm);
  for (int level = 0; level < LEVELS; ++level) {
    const float* src = (level == 0) ? x : residual;
    rvq_level<<<N_PTS / ROWS_PER_BLOCK, 256, 0, stream>>>(
        src, residual,
        cb16 + (size_t)level * KCODES * DIM,
        wnorm + (size_t)level * KCODES,
        cb + (size_t)level * KCODES * DIM,
        probs + (size_t)level * KCODES,
        commit, qsum, codes, level, level == 0 ? 1 : 0);
  }
  rvq_final<<<1, 1024, 0, stream>>>(probs, commit, lossp);
}
